// Model_12945031430931
// MI455X (gfx1250) — compile-verified, matrix operand order checked by reading, not run on hardware
//
#include <hip/hip_runtime.h>
#include <hip/hip_bf16.h>

typedef __attribute__((ext_vector_type(16))) _Float16 v16h;
typedef __attribute__((ext_vector_type(8)))  _Float16 v8h;
typedef __attribute__((ext_vector_type(2)))  _Float16 v2h;
typedef __attribute__((ext_vector_type(8)))  float    v8f;

union V16U { v16h v; v8h h[2]; };

#define TT 64            // time tile per block
#define XS_STRIDE 40     // halves; 80B row stride -> 16B aligned rows

// ---------------------------------------------------------------------------
// WMMA implicit-GEMM conv1d:  out[co,t] = sum_{k,ci} wk[k][co][ci] * in[ci, t+k-PAD]
// in:  f16 [CinPad][T]   wk: f16 [K][CoutPad][CinPad]   out: f32 [Cout][T]
// Block: 256 thr (8 waves). Tile: 256 co x 64 t. Each wave: 32 co x 64 t
// (2 A fragments x 4 time groups = 8 f32 accumulators).
// ---------------------------------------------------------------------------
template <int K, int PAD>
__global__ __launch_bounds__(256) void conv_wmma_kernel(
    const _Float16* __restrict__ in, const _Float16* __restrict__ wk,
    float* __restrict__ out, int CinPad, int Cout, int CoutPad, int T)
{
  constexpr int nTT = TT + K - 1;
  __shared__ __align__(16) _Float16 xs[nTT * XS_STRIDE];
  const int tid  = threadIdx.x;
  const int lane = tid & 31;
  const int wave = tid >> 5;
  const int t0   = blockIdx.x * TT;
  const int co0  = blockIdx.y * 256 + wave * 32;
  const bool active = (co0 < CoutPad);
  const int half0 = (lane & 16) ? 8 : 0;   // K-split: lanes 0-15 vs 16-31
  const int n     = lane & 15;

  v8f acc0[4] = {};
  v8f acc1[4] = {};

  constexpr int nChunkRow = nTT / 2;       // 2-half chunks per cc row
  constexpr int nChunks = 32 * nChunkRow;

  for (int c0 = 0; c0 < CinPad; c0 += 32) {
    // cooperative stage of x tile (transposed: [tt][cc]) with halo, OOB -> 0
    for (int idx = tid; idx < nChunks; idx += 256) {
      int cc = idx / nChunkRow;
      int ch = idx % nChunkRow;
      int tt = ch * 2;
      int t  = t0 - PAD + tt;
      const _Float16* src = in + (size_t)(c0 + cc) * T + t;
      _Float16 v0 = (_Float16)0.0f, v1 = (_Float16)0.0f;
      if (t >= 0 && t + 1 < T) {
        v2h v = *(const v2h*)(src);       // 4B aligned: t0%64==0, PAD even-ish
        v0 = v[0]; v1 = v[1];
      } else {
        if (t >= 0 && t < T) v0 = src[0];
        if (t + 1 >= 0 && t + 1 < T) v1 = src[1];
      }
      xs[tt * XS_STRIDE + cc] = v0;
      xs[(tt + 1) * XS_STRIDE + cc] = v1;
    }
    __syncthreads();
    if (active) {
      // prefetch next ci-chunk of this wave's weight rows into cache
      if (c0 + 32 < CinPad)
        __builtin_prefetch(wk + ((size_t)co0 + n) * CinPad + c0 + 32, 0, 1);
#pragma unroll
      for (int k = 0; k < K; ++k) {
        const _Float16* wp0 =
            wk + ((size_t)k * CoutPad + (co0 + n)) * CinPad + c0 + half0;
        const _Float16* wp1 = wp0 + (size_t)16 * CinPad;
        V16U a0, a1;
        a0.h[0] = *(const v8h*)(wp0);
        a0.h[1] = *(const v8h*)(wp0 + 16);
        a1.h[0] = *(const v8h*)(wp1);
        a1.h[1] = *(const v8h*)(wp1 + 16);
#pragma unroll
        for (int g = 0; g < 4; ++g) {
          const _Float16* xp = xs + (g * 16 + n + k) * XS_STRIDE + half0;
          V16U b;
          b.h[0] = *(const v8h*)(xp);
          b.h[1] = *(const v8h*)(xp + 16);
          acc0[g] = __builtin_amdgcn_wmma_f32_16x16x32_f16(
              false, a0.v, false, b.v, (short)0, acc0[g], false, false);
          acc1[g] = __builtin_amdgcn_wmma_f32_16x16x32_f16(
              false, a1.v, false, b.v, (short)0, acc1[g], false, false);
        }
      }
    }
    __syncthreads();
  }

  if (active) {
    const int m_hi = (lane >> 4) * 8;      // C/D: vgpr i holds M=i (lanes<16) / M=i+8
#pragma unroll
    for (int half = 0; half < 2; ++half) {
      const v8f* acc = half ? acc1 : acc0;
      int cbase = co0 + half * 16;
#pragma unroll
      for (int g = 0; g < 4; ++g) {
        int t = t0 + g * 16 + n;
#pragma unroll
        for (int i = 0; i < 8; ++i) {
          int co = cbase + i + m_hi;
          if (co < Cout) out[(size_t)co * T + t] = acc[g][i];
        }
      }
    }
  }
}

// ---------------------------------------------------------------------------
// conversions / helpers
// ---------------------------------------------------------------------------
__global__ void cvt_x_kernel(const float* __restrict__ src, _Float16* __restrict__ dst,
                             int Creal, int Cpad, int T) {
  int idx = blockIdx.x * blockDim.x + threadIdx.x;
  if (idx >= Cpad * T) return;
  int c = idx / T, t = idx % T;
  dst[(size_t)c * T + t] = (c < Creal) ? (_Float16)src[(size_t)c * T + t] : (_Float16)0.0f;
}

// torch Conv1d weight (O,I,K) -> f16 [K][Opad][Ipad]
__global__ void cvt_w_oik_kernel(const float* __restrict__ src, _Float16* __restrict__ dst,
                                 int CoutR, int CoutP, int CinR, int CinP, int K) {
  int idx = blockIdx.x * blockDim.x + threadIdx.x;
  int N = K * CoutP * CinP;
  if (idx >= N) return;
  int k = idx / (CoutP * CinP);
  int r = idx % (CoutP * CinP);
  int co = r / CinP, ci = r % CinP;
  float v = (ci < CinR && co < CoutR) ? src[((size_t)co * CinR + ci) * K + k] : 0.0f;
  dst[idx] = (_Float16)v;
}

// torch ConvTranspose1d weight (I,O,K) -> equivalent conv weight, flipped k:
// w2[o][i][k] = src[i][o][K-1-k]   -> f16 [K][Opad][Ipad]
__global__ void cvt_w_iok_flip_kernel(const float* __restrict__ src, _Float16* __restrict__ dst,
                                      int CoutR, int CoutP, int CinR, int CinP, int K) {
  int idx = blockIdx.x * blockDim.x + threadIdx.x;
  int N = K * CoutP * CinP;
  if (idx >= N) return;
  int k = idx / (CoutP * CinP);
  int r = idx % (CoutP * CinP);
  int co = r / CinP, ci = r % CinP;
  float v = (ci < CinR && co < CoutR) ? src[((size_t)ci * CoutR + co) * K + (K - 1 - k)] : 0.0f;
  dst[idx] = (_Float16)v;
}

// per-t channel LayerNorm (bias fused) + LeakyReLU, f32 raw -> f16 act
__global__ void ln_lrelu_kernel(const float* __restrict__ raw, const float* __restrict__ bias,
                                const float* __restrict__ g, const float* __restrict__ bt,
                                _Float16* __restrict__ outp, int C, int T) {
  int t = blockIdx.x * blockDim.x + threadIdx.x;
  if (t >= T) return;
  float s = 0.f, s2 = 0.f;
  for (int c = 0; c < C; ++c) {
    float v = raw[(size_t)c * T + t] + bias[c];
    s += v; s2 += v * v;
  }
  float m = s / C;
  float var = s2 / C - m * m;
  float r = rsqrtf(var + 1e-5f);
  for (int c = 0; c < C; ++c) {
    float v = raw[(size_t)c * T + t] + bias[c];
    float y = (v - m) * r * g[c] + bt[c];
    y = (y >= 0.f) ? y : 0.01f * y;
    outp[(size_t)c * T + t] = (_Float16)y;
  }
}

// per-t channel LayerNorm over 2*H channels + GLU -> H f16 rows
__global__ void ln_glu_kernel(const float* __restrict__ raw, const float* __restrict__ bias,
                              const float* __restrict__ g, const float* __restrict__ bt,
                              _Float16* __restrict__ outp, int C, int T) {
  int t = blockIdx.x * blockDim.x + threadIdx.x;
  if (t >= T) return;
  float s = 0.f, s2 = 0.f;
  for (int c = 0; c < C; ++c) {
    float v = raw[(size_t)c * T + t] + bias[c];
    s += v; s2 += v * v;
  }
  float m = s / C;
  float var = s2 / C - m * m;
  float r = rsqrtf(var + 1e-5f);
  int H = C / 2;
  for (int c = 0; c < H; ++c) {
    float va = raw[(size_t)c * T + t] + bias[c];
    va = (va - m) * r * g[c] + bt[c];
    float vb = raw[(size_t)(c + H) * T + t] + bias[c + H];
    vb = (vb - m) * r * g[c + H] + bt[c + H];
    float y = va * (1.0f / (1.0f + expf(-vb)));
    outp[(size_t)c * T + t] = (_Float16)y;
  }
}

__global__ void argmax_kernel(const float* __restrict__ raw, const float* __restrict__ bias,
                              int* __restrict__ ids, int C, int T) {
  int t = blockIdx.x * blockDim.x + threadIdx.x;
  if (t >= T) return;
  int best = 0;
  float bv = raw[t] + bias[0];
  for (int c = 1; c < C; ++c) {
    float v = raw[(size_t)c * T + t] + bias[c];
    if (v > bv) { bv = v; best = c; }   // strict > keeps first max (jnp.argmax)
  }
  ids[t] = best;
}

// blank-drop + run-length dedup + compaction (single block, two block scans)
__global__ __launch_bounds__(1024) void dedup_kernel(const int* __restrict__ ids,
                                                     int* __restrict__ sel,
                                                     float* __restrict__ nvalid, int T) {
  const int tid = threadIdx.x;
  const int SEG = T / 1024;
  const int base = tid * SEG;
  __shared__ int lastnz[1024];
  __shared__ int cnts[1024];

  int ln = -1;
  for (int i = 0; i < SEG; ++i) { int id = ids[base + i]; if (id != 0) ln = id; }
  lastnz[tid] = ln;
  __syncthreads();
  // inclusive scan, op(a,b) = (b>=0 ? b : a)  ("last nonzero so far")
  for (int off = 1; off < 1024; off <<= 1) {
    int u = (tid >= off) ? lastnz[tid - off] : -1;
    __syncthreads();
    if (lastnz[tid] < 0) lastnz[tid] = u;
    __syncthreads();
  }
  int prev0 = (tid == 0) ? -1 : lastnz[tid - 1];

  int prev = prev0, c = 0;
  for (int i = 0; i < SEG; ++i) {
    int id = ids[base + i];
    if (id != 0) { if (id != prev) ++c; prev = id; }
  }
  cnts[tid] = c;
  __syncthreads();
  // inclusive +scan
  for (int off = 1; off < 1024; off <<= 1) {
    int u = (tid >= off) ? cnts[tid - off] : 0;
    __syncthreads();
    cnts[tid] += u;
    __syncthreads();
  }
  int wbase = cnts[tid] - c;
  int total = cnts[1023];

  prev = prev0;
  int o = wbase;
  for (int i = 0; i < SEG; ++i) {
    int id = ids[base + i];
    if (id != 0) { if (id != prev) sel[o++] = id; prev = id; }
  }
  // pad: reference idx fill_value=0 -> sel_ids[pad] = ids[0]
  int fill = ids[0];
  for (int j = total + tid; j < T; j += 1024) sel[j] = fill;
  if (tid == 0) nvalid[0] = (float)total;
}

// row-normalize 128x128 codebook -> f16
__global__ void norm_rows_kernel(const float* __restrict__ src, _Float16* __restrict__ dst) {
  __shared__ float red[128];
  int r = blockIdx.x, c = threadIdx.x;
  float v = src[r * 128 + c];
  red[c] = v * v;
  __syncthreads();
  for (int off = 64; off > 0; off >>= 1) {
    if (c < off) red[c] += red[c + off];
    __syncthreads();
  }
  float rn = rsqrtf(red[0]);
  dst[r * 128 + c] = (_Float16)(v * rn);
}

// row-normalize one speaker row -> f16
__global__ void norm_spk_kernel(const float* __restrict__ spk, const int* __restrict__ y,
                                _Float16* __restrict__ dst) {
  __shared__ float red[128];
  int c = threadIdx.x;
  int row = y[0];
  float v = spk[row * 128 + c];
  red[c] = v * v;
  __syncthreads();
  for (int off = 64; off > 0; off >>= 1) {
    if (c < off) red[c] += red[c + off];
    __syncthreads();
  }
  float rn = rsqrtf(red[0]);
  dst[c] = (_Float16)(v * rn);
}

// act[c][t] = codes[sel[t]][c]   (128 rows)
__global__ void vq_fill_kernel(const _Float16* __restrict__ codes, const int* __restrict__ sel,
                               _Float16* __restrict__ act, int T) {
  int idx = blockIdx.x * blockDim.x + threadIdx.x;
  if (idx >= 128 * T) return;
  int c = idx / T, t = idx % T;
  act[(size_t)c * T + t] = codes[sel[t] * 128 + c];
}

// broadcast y embedding over time into 128 rows at dst
__global__ void fill_y_kernel(const _Float16* __restrict__ yemb, _Float16* __restrict__ dst, int T) {
  int idx = blockIdx.x * blockDim.x + threadIdx.x;
  if (idx >= 128 * T) return;
  int c = idx / T, t = idx % T;
  dst[(size_t)c * T + t] = yemb[c];
}

__global__ void final_bias_kernel(const float* __restrict__ raw, const float* __restrict__ bias,
                                  float* __restrict__ outp, int C, int T) {
  int idx = blockIdx.x * blockDim.x + threadIdx.x;
  if (idx >= C * T) return;
  outp[idx] = raw[idx] + bias[idx / T];
}

// ---------------------------------------------------------------------------
extern "C" void kernel_launch(void* const* d_in, const int* in_sizes, int n_in,
                              void* d_out, int out_size, void* d_ws, size_t ws_size,
                              hipStream_t stream) {
  const int T = 16384;
  const float* x         = (const float*)d_in[0];
  const int*   y         = (const int*)d_in[1];
  const float* enc_w0    = (const float*)d_in[2];
  const float* enc_b0    = (const float*)d_in[3];
  const float* enc_ws    = (const float*)d_in[4];
  const float* enc_bs    = (const float*)d_in[5];
  const float* enc_g     = (const float*)d_in[6];
  const float* enc_bt    = (const float*)d_in[7];
  const float* mlp_w     = (const float*)d_in[8];
  const float* mlp_b     = (const float*)d_in[9];
  const float* codebook  = (const float*)d_in[10];
  const float* spk_table = (const float*)d_in[11];
  const float* d0_w = (const float*)d_in[12];
  const float* d0_b = (const float*)d_in[13];
  const float* d0_g = (const float*)d_in[14];
  const float* d0_bt = (const float*)d_in[15];
  const float* d1_w = (const float*)d_in[16];
  const float* d1_b = (const float*)d_in[17];
  const float* d1_g = (const float*)d_in[18];
  const float* d1_bt = (const float*)d_in[19];
  const float* d2_w = (const float*)d_in[20];
  const float* d2_b = (const float*)d_in[21];

  char* ws = (char*)d_ws;
  size_t off = 0;
  auto take = [&](size_t bytes) -> void* {
    void* p = ws + off;
    off += (bytes + 255) & ~(size_t)255;
    return p;
  };

  _Float16* actA = (_Float16*)take((size_t)1024 * T * 2);
  _Float16* actB = (_Float16*)take((size_t)1024 * T * 2);
  float*    raw  = (float*)take((size_t)1024 * T * 4);
  _Float16* wkE0 = (_Float16*)take((size_t)5 * 512 * 96 * 2);
  _Float16* wkE1 = (_Float16*)take((size_t)5 * 512 * 512 * 2);
  _Float16* wkE2 = (_Float16*)take((size_t)5 * 512 * 512 * 2);
  _Float16* wkE3 = (_Float16*)take((size_t)5 * 512 * 512 * 2);
  _Float16* wkM  = (_Float16*)take((size_t)1 * 128 * 512 * 2);
  _Float16* wkD0 = (_Float16*)take((size_t)5 * 1024 * 256 * 2);
  _Float16* wkD1 = (_Float16*)take((size_t)5 * 1024 * 640 * 2);
  _Float16* wkD2 = (_Float16*)take((size_t)5 * 96 * 640 * 2);
  int* ids = (int*)take((size_t)T * 4);
  int* sel = (int*)take((size_t)T * 4);
  _Float16* codes = (_Float16*)take(128 * 128 * 2);
  _Float16* yemb  = (_Float16*)take(128 * 2);
  float* xhat = (float*)d_out;

  const int THR = 256;
  auto blks = [](int n, int b) { return (n + b - 1) / b; };

  // ---- conversions ----
  cvt_x_kernel<<<blks(96 * T, THR), THR, 0, stream>>>(x, actA, 80, 96, T);
  cvt_w_oik_kernel<<<blks(5 * 512 * 96, THR), THR, 0, stream>>>(enc_w0, wkE0, 512, 512, 80, 96, 5);
  cvt_w_oik_kernel<<<blks(5 * 512 * 512, THR), THR, 0, stream>>>(enc_ws + (size_t)0 * 512 * 512 * 5, wkE1, 512, 512, 512, 512, 5);
  cvt_w_oik_kernel<<<blks(5 * 512 * 512, THR), THR, 0, stream>>>(enc_ws + (size_t)1 * 512 * 512 * 5, wkE2, 512, 512, 512, 512, 5);
  cvt_w_oik_kernel<<<blks(5 * 512 * 512, THR), THR, 0, stream>>>(enc_ws + (size_t)2 * 512 * 512 * 5, wkE3, 512, 512, 512, 512, 5);
  cvt_w_oik_kernel<<<blks(1 * 128 * 512, THR), THR, 0, stream>>>(mlp_w, wkM, 128, 128, 512, 512, 1);
  cvt_w_iok_flip_kernel<<<blks(5 * 1024 * 256, THR), THR, 0, stream>>>(d0_w, wkD0, 1024, 1024, 256, 256, 5);
  cvt_w_iok_flip_kernel<<<blks(5 * 1024 * 640, THR), THR, 0, stream>>>(d1_w, wkD1, 1024, 1024, 640, 640, 5);
  cvt_w_iok_flip_kernel<<<blks(5 * 96 * 640, THR), THR, 0, stream>>>(d2_w, wkD2, 80, 96, 640, 640, 5);
  norm_rows_kernel<<<128, 128, 0, stream>>>(codebook, codes);
  norm_spk_kernel<<<1, 128, 0, stream>>>(spk_table, y, yemb);

  // ---- encoder: 4x (conv k5 p2 -> chLN -> lrelu) ----
  conv_wmma_kernel<5, 2><<<dim3(T / TT, blks(512, 256)), THR, 0, stream>>>(actA, wkE0, raw, 96, 512, 512, T);
  ln_lrelu_kernel<<<blks(T, THR), THR, 0, stream>>>(raw, enc_b0, enc_g + 0 * 512, enc_bt + 0 * 512, actB, 512, T);
  conv_wmma_kernel<5, 2><<<dim3(T / TT, blks(512, 256)), THR, 0, stream>>>(actB, wkE1, raw, 512, 512, 512, T);
  ln_lrelu_kernel<<<blks(T, THR), THR, 0, stream>>>(raw, enc_bs + 0 * 512, enc_g + 1 * 512, enc_bt + 1 * 512, actA, 512, T);
  conv_wmma_kernel<5, 2><<<dim3(T / TT, blks(512, 256)), THR, 0, stream>>>(actA, wkE2, raw, 512, 512, 512, T);
  ln_lrelu_kernel<<<blks(T, THR), THR, 0, stream>>>(raw, enc_bs + 1 * 512, enc_g + 2 * 512, enc_bt + 2 * 512, actB, 512, T);
  conv_wmma_kernel<5, 2><<<dim3(T / TT, blks(512, 256)), THR, 0, stream>>>(actB, wkE3, raw, 512, 512, 512, T);
  ln_lrelu_kernel<<<blks(T, THR), THR, 0, stream>>>(raw, enc_bs + 2 * 512, enc_g + 3 * 512, enc_bt + 3 * 512, actA, 512, T);

  // ---- 1x1 projection to z_num, argmax, dedup ----
  conv_wmma_kernel<1, 0><<<dim3(T / TT, blks(128, 256)), THR, 0, stream>>>(actA, wkM, raw, 512, 128, 128, T);
  argmax_kernel<<<blks(T, THR), THR, 0, stream>>>(raw, mlp_b, ids, 128, T);
  dedup_kernel<<<1, 1024, 0, stream>>>(ids, sel, xhat + (size_t)80 * T, T);

  // ---- VQ lookup + speaker concat -> decoder input (256 rows) ----
  vq_fill_kernel<<<blks(128 * T, THR), THR, 0, stream>>>(codes, sel, actA, T);
  fill_y_kernel<<<blks(128 * T, THR), THR, 0, stream>>>(yemb, actA + (size_t)128 * T, T);

  // ---- decoder d0: convT(k5,p2) -> chLN(1024) -> GLU -> 512 + y(128) ----
  conv_wmma_kernel<5, 2><<<dim3(T / TT, blks(1024, 256)), THR, 0, stream>>>(actA, wkD0, raw, 256, 1024, 1024, T);
  ln_glu_kernel<<<blks(T, THR), THR, 0, stream>>>(raw, d0_b, d0_g, d0_bt, actB, 1024, T);
  fill_y_kernel<<<blks(128 * T, THR), THR, 0, stream>>>(yemb, actB + (size_t)512 * T, T);

  // ---- decoder d1 ----
  conv_wmma_kernel<5, 2><<<dim3(T / TT, blks(1024, 256)), THR, 0, stream>>>(actB, wkD1, raw, 640, 1024, 1024, T);
  ln_glu_kernel<<<blks(T, THR), THR, 0, stream>>>(raw, d1_b, d1_g, d1_bt, actA, 1024, T);
  fill_y_kernel<<<blks(128 * T, THR), THR, 0, stream>>>(yemb, actA + (size_t)512 * T, T);

  // ---- decoder d2 (plain convT) + bias -> xhat ----
  conv_wmma_kernel<5, 2><<<dim3(T / TT, blks(96, 256)), THR, 0, stream>>>(actA, wkD2, raw, 640, 80, 96, T);
  final_bias_kernel<<<blks(80 * T, THR), THR, 0, stream>>>(raw, d2_b, xhat, 80, T);
}
